// N2V_GAT_75866302317080
// MI455X (gfx1250) — matrix operand order repaired and checked by reading, not compiled, not run on hardware
//
#include <hip/hip_runtime.h>
#include <hip/hip_bf16.h>

// ---------------------------------------------------------------------------
// GATv2 (2 layers, heads=4 then 1) + BatchNorm + ELU + Linear classifier.
// Dense GEMMs use v_wmma_f32_16x16x32_bf16 (bf16 inputs, fp32 accumulate),
// fully unrolled over K with NT column-tiles per wave (A-fragment reuse).
// Edge softmax/scatter uses fp32 global atomics (L2-resident working set).
// ---------------------------------------------------------------------------

typedef __attribute__((ext_vector_type(16))) __bf16 v16bf;
typedef __attribute__((ext_vector_type(8)))  float  v8f;

#define NEG_SLOPE 0.2f
#define BN_EPS 1e-5f

__device__ __forceinline__ unsigned short f2bf(float f) {
    // round-to-nearest-even fp32 -> bf16
    unsigned int u = __float_as_uint(f);
    unsigned int r = u + 0x7fffu + ((u >> 16) & 1u);
    return (unsigned short)(r >> 16);
}

union BfFrag {
    v16bf v;
    unsigned short u[16];
};

__device__ __forceinline__ unsigned int fkey(float f) {
    // monotone mapping float -> uint so unsigned atomicMax == float max
    unsigned int u = __float_as_uint(f);
    return (u & 0x80000000u) ? ~u : (u | 0x80000000u);
}
__device__ __forceinline__ float funkey(unsigned int k) {
    unsigned int u = (k & 0x80000000u) ? (k ^ 0x80000000u) : ~k;
    return __uint_as_float(u);
}

__device__ __forceinline__ void atomic_add_f32(float* p, float v) {
    __hip_atomic_fetch_add(p, v, __ATOMIC_RELAXED, __HIP_MEMORY_SCOPE_AGENT);
}

// ---------------------------------------------------------------------------
// WMMA GEMM: C[M x Ncols] = A[M x K] @ B[K x Ncols].
// One wave computes a 16 x (16*NT) strip: A fragment loaded once per K-step,
// NT WMMAs issued against it. K is compile-time -> K-loop fully unrolled.
// MODE 0: C = A@B (+bias if non-null).  MODE 1: C += A@B.
// Requirements (all call sites satisfy): M%16==0 (M=50000=16*3125),
// K%32==0, Ncols % (16*NT) == 0, ldb == Ncols.
// A/B fragments use contiguous K = 16*half + s (same in both); f32 C/D per CDNA5 ISA 7.12.2.
// ---------------------------------------------------------------------------
template <int K, int NT, int MODE>
__global__ __launch_bounds__(32)
void wmma_gemm_bf16_kernel(const float* __restrict__ A,
                           const float* __restrict__ B,
                           float* __restrict__ C,
                           const float* __restrict__ bias,
                           int Ncols)
{
    const int lane = threadIdx.x & 31;
    const int half = lane >> 4;     // K-half selector
    const int r    = lane & 15;     // A: row within tile; B/C: column within tile
    const int m0   = blockIdx.x << 4;
    const int n0   = blockIdx.y * (16 * NT);

    v8f c[NT];
#pragma unroll
    for (int t = 0; t < NT; ++t) c[t] = (v8f){};

    const float* arow = A + (size_t)(m0 + r) * K;

#pragma unroll
    for (int k0 = 0; k0 < K; k0 += 32) {
        // ---- A fragment: row (m0+r), 16 K-values per lane ----
        BfFrag a;
#pragma unroll
        for (int i = 0; i < 8; ++i) {
            // lanes 0-15: K = {0..15}; lanes 16-31: K = {16..31}
            int kb = 2 * i + 16 * half;
            float2 f = *(const float2*)(arow + k0 + kb);
            a.u[2 * i]     = f2bf(f.x);
            a.u[2 * i + 1] = f2bf(f.y);
        }
#pragma unroll
        for (int t = 0; t < NT; ++t) {
            // ---- B fragment: column (n0+16t+r), K = k0 + 16*half + 2i + {0,1}
            BfFrag b;
            const float* bcol =
                B + (size_t)(k0 + 16 * half) * Ncols + (n0 + 16 * t + r);
#pragma unroll
            for (int i = 0; i < 8; ++i) {
                b.u[2 * i]     = f2bf(bcol[(2 * i) * Ncols]);
                b.u[2 * i + 1] = f2bf(bcol[(2 * i + 1) * Ncols]);
            }
            // (neg_a, A, neg_b, B, c_mod, C, reuse_a, reuse_b)
            c[t] = __builtin_amdgcn_wmma_f32_16x16x32_bf16(
                       false, a.v, false, b.v, (short)0, c[t], false, false);
        }
    }

    // ---- store: element v -> row m0 + v + 8*half, col n0 + 16t + r ----
#pragma unroll
    for (int t = 0; t < NT; ++t) {
        const int nc = n0 + 16 * t + r;
        float* crow = C + (size_t)(m0 + 8 * half) * Ncols + nc;
        if (MODE == 0) {
            const float badd = (bias != nullptr) ? bias[nc] : 0.0f;
#pragma unroll
            for (int v = 0; v < 8; ++v)
                crow[(size_t)v * Ncols] = c[t][v] + badd;
        } else {
#pragma unroll
            for (int v = 0; v < 8; ++v)
                crow[(size_t)v * Ncols] += c[t][v];
        }
    }
}

// ---------------------------------------------------------------------------
// Edge pass 1: attention logits + per-(dst,head) running max.
// One thread per (edge, head); self-loops are edges [E_orig, Etot).
// ---------------------------------------------------------------------------
template <int H>
__global__ void edge_logits_kernel(const float* __restrict__ xl,
                                   const float* __restrict__ xr,
                                   const float* __restrict__ att,
                                   const int* __restrict__ src,
                                   const int* __restrict__ dst,
                                   float* __restrict__ logits,
                                   unsigned int* __restrict__ maxkey,
                                   int E_orig, int Etot)
{
    int t = blockIdx.x * blockDim.x + threadIdx.x;
    if (t >= Etot * H) return;
    const int e = t / H;
    const int h = t % H;
    int s, d;
    if (e < E_orig) { s = src[e]; d = dst[e]; }
    else            { s = d = e - E_orig; }

    const float* xls = xl + (size_t)s * (H * 32) + h * 32;
    const float* xrd = xr + (size_t)d * (H * 32) + h * 32;
    const float* ah  = att + h * 32;

    float acc = 0.0f;
#pragma unroll
    for (int c = 0; c < 32; c += 4) {
        float4 a = *(const float4*)(xls + c);
        float4 b = *(const float4*)(xrd + c);
        float4 w = *(const float4*)(ah + c);
        float g;
        g = a.x + b.x; g = (g > 0.0f) ? g : NEG_SLOPE * g; acc += g * w.x;
        g = a.y + b.y; g = (g > 0.0f) ? g : NEG_SLOPE * g; acc += g * w.y;
        g = a.z + b.z; g = (g > 0.0f) ? g : NEG_SLOPE * g; acc += g * w.z;
        g = a.w + b.w; g = (g > 0.0f) ? g : NEG_SLOPE * g; acc += g * w.w;
    }
    logits[(size_t)e * H + h] = acc;
    atomicMax(&maxkey[(size_t)d * H + h], fkey(acc));
}

// ---------------------------------------------------------------------------
// Edge pass 2: w = exp(logit - max[dst]); denom[dst] += w;
//              acc[dst] += w * xl[src]  (32 fp32 atomic adds)
// ---------------------------------------------------------------------------
template <int H>
__global__ void edge_scatter_kernel(const float* __restrict__ xl,
                                    const float* __restrict__ logits,
                                    const unsigned int* __restrict__ maxkey,
                                    const int* __restrict__ src,
                                    const int* __restrict__ dst,
                                    float* __restrict__ denom,
                                    float* __restrict__ acc,
                                    int E_orig, int Etot)
{
    int t = blockIdx.x * blockDim.x + threadIdx.x;
    if (t >= Etot * H) return;
    const int e = t / H;
    const int h = t % H;
    int s, d;
    if (e < E_orig) { s = src[e]; d = dst[e]; }
    else            { s = d = e - E_orig; }

    const float lg = logits[(size_t)e * H + h];
    const float mx = funkey(maxkey[(size_t)d * H + h]);
    const float w  = __expf(lg - mx);

    atomic_add_f32(&denom[(size_t)d * H + h], w);

    const float* xls  = xl + (size_t)s * (H * 32) + h * 32;
    float*       outp = acc + (size_t)d * (H * 32) + h * 32;
#pragma unroll
    for (int c = 0; c < 32; ++c)
        atomic_add_f32(&outp[c], w * xls[c]);
}

// ---------------------------------------------------------------------------
// conv1 finalize: acc = acc/denom + b1 (in place), fused BN batch statistics.
// grid = P blocks of 128 threads; block p handles nodes p, p+P, ...
// ---------------------------------------------------------------------------
__global__ void finalize1_kernel(float* __restrict__ acc,
                                 const float* __restrict__ denom,
                                 const float* __restrict__ b1,
                                 float* __restrict__ bnsum,
                                 float* __restrict__ bnsumsq,
                                 int Nn)
{
    const int ch = threadIdx.x;     // 0..127
    const int h  = ch >> 5;
    const float bias = b1[ch];
    float s = 0.0f, sq = 0.0f;
    for (int n = blockIdx.x; n < Nn; n += gridDim.x) {
        size_t i = (size_t)n * 128 + ch;
        float v = acc[i] / denom[(size_t)n * 4 + h] + bias;
        acc[i] = v;
        s  += v;
        sq += v * v;
    }
    atomic_add_f32(&bnsum[ch], s);
    atomic_add_f32(&bnsumsq[ch], sq);
}

__global__ void bn_scale_kernel(const float* __restrict__ bnsum,
                                const float* __restrict__ bnsumsq,
                                const float* __restrict__ gamma,
                                const float* __restrict__ beta,
                                float* __restrict__ scale,
                                float* __restrict__ shift,
                                int Nn)
{
    const int ch = threadIdx.x;     // 0..127
    const float inv  = 1.0f / (float)Nn;
    const float mean = bnsum[ch] * inv;
    const float var  = bnsumsq[ch] * inv - mean * mean;
    const float sc   = gamma[ch] * rsqrtf(var + BN_EPS);
    scale[ch] = sc;
    shift[ch] = beta[ch] - mean * sc;
}

__global__ void bn_apply_elu_kernel(float* __restrict__ h,
                                    const float* __restrict__ scale,
                                    const float* __restrict__ shift,
                                    int total)
{
    int i = blockIdx.x * blockDim.x + threadIdx.x;
    if (i >= total) return;
    const int ch = i & 127;
    float v = h[i] * scale[ch] + shift[ch];
    h[i] = (v > 0.0f) ? v : (__expf(v) - 1.0f);
}

// conv2 finalize: h2 = elu(acc/denom + b2)
__global__ void finalize2_kernel(const float* __restrict__ acc,
                                 const float* __restrict__ denom,
                                 const float* __restrict__ b2,
                                 float* __restrict__ h2,
                                 int total)
{
    int i = blockIdx.x * blockDim.x + threadIdx.x;
    if (i >= total) return;
    const int ch   = i & 31;
    const int node = i >> 5;
    float v = acc[i] / denom[node] + b2[ch];
    h2[i] = (v > 0.0f) ? v : (__expf(v) - 1.0f);
}

// ---------------------------------------------------------------------------
extern "C" void kernel_launch(void* const* d_in, const int* in_sizes, int n_in,
                              void* d_out, int out_size, void* d_ws, size_t ws_size,
                              hipStream_t stream)
{
    const float* x    = (const float*)d_in[0];
    const int*   ei   = (const int*)d_in[1];
    const float* W1l  = (const float*)d_in[2];
    const float* W1r  = (const float*)d_in[3];
    const float* att1 = (const float*)d_in[4];
    const float* b1   = (const float*)d_in[5];
    const float* bng  = (const float*)d_in[6];
    const float* bnb  = (const float*)d_in[7];
    const float* W2l  = (const float*)d_in[8];
    const float* W2r  = (const float*)d_in[9];
    const float* att2 = (const float*)d_in[10];
    const float* b2   = (const float*)d_in[11];
    const float* Wc   = (const float*)d_in[12];
    const float* bc   = (const float*)d_in[13];
    float* out = (float*)d_out;

    const int Nn   = in_sizes[0] / 128;   // 50000 (multiple of 16)
    const int E    = in_sizes[1] / 2;     // 800000
    const int Etot = E + Nn;              // + self-loops
    const int* src = ei;
    const int* dst = ei + E;

    // ---- workspace layout (floats) ----
    float* xl1   = (float*)d_ws;                    // N*128
    float* xr1   = xl1 + (size_t)Nn * 128;          // N*128
    float* accb  = xr1 + (size_t)Nn * 128;          // N*128 (acc1 -> h1 in place)
    float* logit = accb + (size_t)Nn * 128;         // Etot*4
    unsigned int* maxk = (unsigned int*)(logit + (size_t)Etot * 4);  // N*4
    float* denom = (float*)(maxk + (size_t)Nn * 4); // N*4
    float* bnsum   = denom + (size_t)Nn * 4;        // 128
    float* bnsumsq = bnsum + 128;                   // 128
    float* bnscale = bnsumsq + 128;                 // 128
    float* bnshift = bnscale + 128;                 // 128
    // conv2 buffers alias the (dead after conv1) xl1 region:
    float* xl2  = xl1;                              // N*32
    float* xr2  = xl1 + (size_t)Nn * 32;            // N*32
    float* acc2 = xl1 + (size_t)Nn * 64;            // N*32
    float* h2   = xl1 + (size_t)Nn * 96;            // N*32

    const dim3 wave(32);
    const int mtiles = Nn / 16;

    // ---- conv1 projections: xl1 = x@W1l, xr1 = x@W1r (WMMA bf16) ----
    // 16x64 strip per wave: grid.y = 128/64 = 2
    wmma_gemm_bf16_kernel<128, 4, 0><<<dim3(mtiles, 2), wave, 0, stream>>>(
        x, W1l, xl1, nullptr, 128);
    wmma_gemm_bf16_kernel<128, 4, 0><<<dim3(mtiles, 2), wave, 0, stream>>>(
        x, W1r, xr1, nullptr, 128);

    // ---- zero accumulators (stream-ordered, graph-capturable) ----
    hipMemsetAsync(maxk,  0, (size_t)Nn * 4 * sizeof(unsigned int), stream);
    hipMemsetAsync(denom, 0, (size_t)Nn * 4 * sizeof(float), stream);
    hipMemsetAsync(accb,  0, (size_t)Nn * 128 * sizeof(float), stream);
    hipMemsetAsync(bnsum, 0, 2 * 128 * sizeof(float), stream);

    // ---- conv1 edge softmax + scatter ----
    const int t1 = Etot * 4;
    edge_logits_kernel<4><<<(t1 + 255) / 256, 256, 0, stream>>>(
        xl1, xr1, att1, src, dst, logit, maxk, E, Etot);
    edge_scatter_kernel<4><<<(t1 + 255) / 256, 256, 0, stream>>>(
        xl1, logit, maxk, src, dst, denom, accb, E, Etot);

    // ---- finalize + BN + ELU ----
    finalize1_kernel<<<512, 128, 0, stream>>>(accb, denom, b1, bnsum, bnsumsq, Nn);
    bn_scale_kernel<<<1, 128, 0, stream>>>(bnsum, bnsumsq, bng, bnb, bnscale, bnshift, Nn);
    bn_apply_elu_kernel<<<(Nn * 128 + 255) / 256, 256, 0, stream>>>(
        accb, bnscale, bnshift, Nn * 128);

    // ---- conv2 projections: xl2 = h1@W2l, xr2 = h1@W2r ----
    // 16x32 strip per wave: grid.y = 32/32 = 1
    wmma_gemm_bf16_kernel<128, 2, 0><<<dim3(mtiles, 1), wave, 0, stream>>>(
        accb, W2l, xl2, nullptr, 32);
    wmma_gemm_bf16_kernel<128, 2, 0><<<dim3(mtiles, 1), wave, 0, stream>>>(
        accb, W2r, xr2, nullptr, 32);

    hipMemsetAsync(maxk,  0, (size_t)Nn * sizeof(unsigned int), stream);
    hipMemsetAsync(denom, 0, (size_t)Nn * sizeof(float), stream);
    hipMemsetAsync(acc2,  0, (size_t)Nn * 32 * sizeof(float), stream);

    // ---- conv2 edge softmax + scatter (1 head) ----
    edge_logits_kernel<1><<<(Etot + 255) / 256, 256, 0, stream>>>(
        xl2, xr2, att2, src, dst, logit, maxk, E, Etot);
    edge_scatter_kernel<1><<<(Etot + 255) / 256, 256, 0, stream>>>(
        xl2, logit, maxk, src, dst, denom, acc2, E, Etot);
    finalize2_kernel<<<(Nn * 32 + 255) / 256, 256, 0, stream>>>(
        acc2, denom, b2, h2, Nn * 32);

    // ---- classifier: out = [h2 | x] @ Wc + bc (Wc rows 0..31 for h2) ----
    // 16x64 strip per wave: grid.y = 64/64 = 1
    wmma_gemm_bf16_kernel<32, 4, 0><<<dim3(mtiles, 1), wave, 0, stream>>>(
        h2, Wc, out, bc, 64);
    wmma_gemm_bf16_kernel<128, 4, 1><<<dim3(mtiles, 1), wave, 0, stream>>>(
        x, Wc + 32 * 64, out, nullptr, 64);
}
